// _NonLocalBlockND_65970697667349
// MI455X (gfx1250) — matrix operand order repaired and checked by reading, not compiled, not run on hardware
//
#include <hip/hip_runtime.h>

typedef __attribute__((ext_vector_type(16))) _Float16 v16h;
typedef __attribute__((ext_vector_type(8)))  _Float16 v8h;
typedef __attribute__((ext_vector_type(8)))  float    v8f;

#define BB  2
#define CC  64
#define ICC 32
#define HH  80
#define WWD 80
#define NN  (HH * WWD)

// ---------------------------------------------------------------------------
// Kernel 1: shared 3x3 conv -> theta (from x) and phi (from x_t), both stored
// as (N, 32) f16 row-major; 1x1 conv g (from x_t) stored transposed (32, N).
// Layouts are chosen so attention WMMA operands are contiguous 16B lane loads.
// ---------------------------------------------------------------------------
__global__ __launch_bounds__(256) void nlb_prep(
    const float* __restrict__ x, const float* __restrict__ xt,
    const float* __restrict__ gw, const float* __restrict__ gb,
    const float* __restrict__ sw, const float* __restrict__ sb,
    _Float16* __restrict__ thetaH, _Float16* __restrict__ phiH,
    _Float16* __restrict__ gT)
{
    int idx = blockIdx.x * blockDim.x + threadIdx.x;
    if (idx >= BB * ICC * NN) return;
    int hw = idx % NN;
    int t  = idx / NN;
    int o  = t % ICC;
    int b  = t / ICC;
    int h = hw / WWD, w = hw % WWD;

    const float* xb  = x  + (size_t)b * CC * NN;
    const float* xtb = xt + (size_t)b * CC * NN;

    float at = sb[o], ap = sb[o];
    for (int c = 0; c < CC; ++c) {
        const float* swp = sw + ((size_t)o * CC + c) * 9;
        const float* xc  = xb  + (size_t)c * NN;
        const float* xtc = xtb + (size_t)c * NN;
        #pragma unroll
        for (int kh = 0; kh < 3; ++kh) {
            int hh = h + kh - 1;
            if ((unsigned)hh >= (unsigned)HH) continue;
            #pragma unroll
            for (int kw = 0; kw < 3; ++kw) {
                int ww = w + kw - 1;
                if ((unsigned)ww >= (unsigned)WWD) continue;
                float wt = swp[kh * 3 + kw];
                at += wt * xc[hh * WWD + ww];
                ap += wt * xtc[hh * WWD + ww];
            }
        }
    }
    float ag = gb[o];
    for (int c = 0; c < CC; ++c)
        ag += gw[o * CC + c] * xtb[(size_t)c * NN + hw];

    thetaH[((size_t)b * NN + hw) * ICC + o] = (_Float16)at;
    phiH  [((size_t)b * NN + hw) * ICC + o] = (_Float16)ap;
    gT    [((size_t)b * ICC + o) * NN + hw] = (_Float16)ag;
}

// ---------------------------------------------------------------------------
// Kernel 2: flash attention, one wave (32 lanes) per 16-query tile.
// Scores computed transposed (keys=M, queries=N): each lane's D registers hold
// a single query's scores, so online softmax needs one shfl_xor(16) only.
// Per 32-key step: 2 score WMMAs + 2 PV WMMAs (f32 accumulators).
// ---------------------------------------------------------------------------
__global__ __launch_bounds__(128) void nlb_attn(
    const _Float16* __restrict__ thetaH, const _Float16* __restrict__ phiH,
    const _Float16* __restrict__ gT, float* __restrict__ yT)
{
    const int lane = threadIdx.x & 31;
    const int wid  = threadIdx.x >> 5;
    const int wg   = blockIdx.x * 4 + wid;          // 800 waves total
    const int b    = wg / (NN / 16);
    const int qt   = wg % (NN / 16);
    const int qbase = qt * 16;
    const bool lo  = lane < 16;
    const int l16  = lane & 15;

    const _Float16* th = thetaH + (size_t)b * NN * ICC;
    const _Float16* ph = phiH   + (size_t)b * NN * ICC;
    const _Float16* gp = gT     + (size_t)b * ICC * NN;
    float*          yp = yT     + (size_t)b * ICC * NN;

    // B operand (theta^T): column N = query, K = channel. Loaded once.
    v16h Bth;
    {
        const _Float16* tq = th + (size_t)(qbase + l16) * ICC + (lo ? 0 : 8);
        v8h a0 = *(const v8h*)(tq);
        v8h a1 = *(const v8h*)(tq + 16);
        #pragma unroll
        for (int i = 0; i < 8; ++i) { Bth[i] = a0[i]; Bth[8 + i] = a1[i]; }
    }

    float m = -1e30f, l = 0.0f;
    v8f acc0 = {};   // y^T rows c = 0..15
    v8f acc1 = {};   // y^T rows c = 16..31

    for (int kb = 0; kb < NN; kb += 32) {
        // A operands: phi rows (keys kb..kb+15 and kb+16..kb+31), K = channel
        v16h A0, A1;
        {
            const _Float16* p0 = ph + (size_t)(kb + l16) * ICC;
            const _Float16* p1 = ph + (size_t)(kb + 16 + l16) * ICC;
            v8h x0 = *(const v8h*)(p0 + (lo ? 0 : 8));
            v8h x1 = *(const v8h*)(p0 + (lo ? 16 : 24));
            v8h y0 = *(const v8h*)(p1 + (lo ? 0 : 8));
            v8h y1 = *(const v8h*)(p1 + (lo ? 16 : 24));
            #pragma unroll
            for (int i = 0; i < 8; ++i) {
                A0[i] = x0[i]; A0[8 + i] = x1[i];
                A1[i] = y0[i]; A1[8 + i] = y1[i];
            }
        }
        v8f zc = {};
        v8f s0 = __builtin_amdgcn_wmma_f32_16x16x32_f16(false, A0, false, Bth,
                                                        (short)0, zc, false, false);
        v8f s1 = __builtin_amdgcn_wmma_f32_16x16x32_f16(false, A1, false, Bth,
                                                        (short)0, zc, false, false);

        // Online softmax (per query = per lane; partner lane = lane ^ 16)
        float tm = s0[0];
        #pragma unroll
        for (int j = 0; j < 8; ++j) { tm = fmaxf(tm, s0[j]); tm = fmaxf(tm, s1[j]); }
        tm = fmaxf(tm, __shfl_xor(tm, 16, 32));
        float mnew  = fmaxf(m, tm);
        float alpha = __expf(m - mnew);
        float pz0[8], pz1[8];
        float ts = 0.0f;
        #pragma unroll
        for (int j = 0; j < 8; ++j) {
            pz0[j] = __expf(s0[j] - mnew);
            pz1[j] = __expf(s1[j] - mnew);
            ts += pz0[j] + pz1[j];
        }
        ts += __shfl_xor(ts, 16, 32);
        l = l * alpha + ts;
        m = mnew;
        #pragma unroll
        for (int j = 0; j < 8; ++j) { acc0[j] *= alpha; acc1[j] *= alpha; }

        // Repack P (two 16x16 f32 D tiles) into a 32x16 f16 B operand:
        // lane<16 keeps keys 0..7 (own) + 8..15 (partner); lane>=16 gets
        // keys 16..23 (partner) + 24..31 (own).
        v16h Bp;
        #pragma unroll
        for (int j = 0; j < 8; ++j) {
            float p0p = __shfl_xor(pz0[j], 16, 32);
            float p1p = __shfl_xor(pz1[j], 16, 32);
            Bp[j]     = (_Float16)(lo ? pz0[j] : p1p);
            Bp[8 + j] = (_Float16)(lo ? p0p   : pz1[j]);
        }

        // A operands from gT: row M = channel, K = key within tile
        v16h Ag0, Ag1;
        {
            const _Float16* g0 = gp + (size_t)l16 * NN + kb;
            const _Float16* g1 = gp + (size_t)(16 + l16) * NN + kb;
            v8h x0 = *(const v8h*)(g0 + (lo ? 0 : 16));
            v8h x1 = *(const v8h*)(g0 + (lo ? 8 : 24));
            v8h y0 = *(const v8h*)(g1 + (lo ? 0 : 16));
            v8h y1 = *(const v8h*)(g1 + (lo ? 8 : 24));
            #pragma unroll
            for (int i = 0; i < 8; ++i) {
                Ag0[i] = x0[i]; Ag0[8 + i] = x1[i];
                Ag1[i] = y0[i]; Ag1[8 + i] = y1[i];
            }
        }
        acc0 = __builtin_amdgcn_wmma_f32_16x16x32_f16(false, Ag0, false, Bp,
                                                      (short)0, acc0, false, false);
        acc1 = __builtin_amdgcn_wmma_f32_16x16x32_f16(false, Ag1, false, Bp,
                                                      (short)0, acc1, false, false);
    }

    // Normalize and write y^T (32, N) f32. Lane holds query q = qbase+l16,
    // channels (lo ? 0..7 : 8..15) (+16 for acc1).
    float inv = 1.0f / l;
    int q  = qbase + l16;
    int cb = lo ? 0 : 8;
    #pragma unroll
    for (int j = 0; j < 8; ++j) {
        yp[(size_t)(cb + j) * NN + q]      = acc0[j] * inv;
        yp[(size_t)(16 + cb + j) * NN + q] = acc1[j] * inv;
    }
}

// ---------------------------------------------------------------------------
// Kernel 3: W (64x32 1x1 conv) + bias + residual, fp32.
// ---------------------------------------------------------------------------
__global__ __launch_bounds__(256) void nlb_epi(
    const float* __restrict__ yT, const float* __restrict__ Ww,
    const float* __restrict__ Wb, const float* __restrict__ x,
    float* __restrict__ out)
{
    int idx = blockIdx.x * blockDim.x + threadIdx.x;
    if (idx >= BB * CC * NN) return;
    int hw = idx % NN;
    int t  = idx / NN;
    int o  = t % CC;
    int b  = t / CC;
    float a = Wb[o];
    const float* yp = yT + (size_t)b * ICC * NN;
    #pragma unroll
    for (int c = 0; c < ICC; ++c)
        a += Ww[o * ICC + c] * yp[(size_t)c * NN + hw];
    out[idx] = a + x[idx];
}

// ---------------------------------------------------------------------------
extern "C" void kernel_launch(void* const* d_in, const int* in_sizes, int n_in,
                              void* d_out, int out_size, void* d_ws, size_t ws_size,
                              hipStream_t stream) {
    (void)in_sizes; (void)n_in; (void)out_size; (void)ws_size;
    const float* x  = (const float*)d_in[0];
    const float* xt = (const float*)d_in[1];
    const float* gw = (const float*)d_in[2];
    const float* gb = (const float*)d_in[3];
    const float* Ww = (const float*)d_in[4];
    const float* Wb = (const float*)d_in[5];
    const float* sw = (const float*)d_in[6];
    const float* sb = (const float*)d_in[7];
    float* out = (float*)d_out;

    char* ws = (char*)d_ws;
    const size_t half_sz = (size_t)BB * NN * ICC * sizeof(_Float16); // 819,200 B
    _Float16* thetaH = (_Float16*)(ws);
    _Float16* phiH   = (_Float16*)(ws + half_sz);
    _Float16* gT     = (_Float16*)(ws + 2 * half_sz);
    float*    yT     = (float*)   (ws + 3 * half_sz);               // 1,638,400 B

    nlb_prep<<<(BB * ICC * NN + 255) / 256, 256, 0, stream>>>(
        x, xt, gw, gb, sw, sb, thetaH, phiH, gT);
    nlb_attn<<<(BB * (NN / 16)) / 4, 128, 0, stream>>>(thetaH, phiH, gT, yT);
    nlb_epi<<<(BB * CC * NN + 255) / 256, 256, 0, stream>>>(yT, Ww, Wb, x, out);
}
